// TemporalMultiHeadAttention_66537633350004
// MI455X (gfx1250) — compile-verified, matrix operand order checked by reading, not run on hardware
//
#include <hip/hip_runtime.h>
#include <hip/hip_bf16.h>
#include <cstdint>
#include <cstddef>

typedef __attribute__((ext_vector_type(16))) __bf16 v16bf;
typedef __attribute__((ext_vector_type(8)))  __bf16 bf16x8;
typedef __attribute__((ext_vector_type(4)))  __bf16 bf16x4;
typedef __attribute__((ext_vector_type(8)))  float  v8f;
typedef __attribute__((ext_vector_type(4)))  float  f32x4;

#define B_   4
#define N_   1024
#define D_   1024
#define H_   16
#define DK_  64
#define MRP_ 32
#define BH_  (B_ * H_)          // 64
#define M_   (B_ * N_)          // 4096
#define NR_  (2 * MRP_ + 1)     // 65

// LDS leading dims (bf16 elems): multiple of 8 keeps the two 16-byte
// fragment chunks per lane 16B-aligned; odd-in-banks to spread conflicts.
#define LDT32 40   // for 32-wide K panels
#define LDT64 72   // for 64-wide K panels

// ---------------------------------------------------------------------------
// WMMA helpers: 16x16x32 bf16 -> f32.
// A-frag (16x32, ISA layout): lane&15 = M row, lane>>4 = K-half; per lane two
// contiguous 8-elem chunks at K = k0+khalf*8 and K = k0+16+khalf*8.
// B-frag assumed mirror-symmetric (N <-> M): both tiles staged row-major
// [row][K] in LDS and share the loader.
// ---------------------------------------------------------------------------
__device__ __forceinline__ v16bf load_frag(const __bf16* t, int row0, int k0, int ld) {
  const int lane  = threadIdx.x & 31;
  const int r     = lane & 15;
  const int khalf = lane >> 4;
  const __bf16* p = t + (size_t)(row0 + r) * ld + k0 + khalf * 8;
  union { v16bf v; bf16x8 h[2]; } u;
  u.h[0] = *(const bf16x8*)(p);
  u.h[1] = *(const bf16x8*)(p + 16);
  return u.v;
}

__device__ __forceinline__ v8f wmma_bf16(v16bf a, v16bf b, v8f c) {
  return __builtin_amdgcn_wmma_f32_16x16x32_bf16(
      /*neg_a=*/false, a, /*neg_b=*/false, b,
      /*c_mod=*/(short)0, c, /*reuse_a=*/false, /*reuse_b=*/false);
}

// 4-element load (16B for f32 with convert, 8B for bf16 pass-through).
__device__ __forceinline__ bf16x4 ld4_cvt(const float* p) {
  const f32x4 v = *(const f32x4*)p;
  bf16x4 h;
  h[0] = (__bf16)v[0]; h[1] = (__bf16)v[1];
  h[2] = (__bf16)v[2]; h[3] = (__bf16)v[3];
  return h;
}
__device__ __forceinline__ bf16x4 ld4_cvt(const __bf16* p) {
  return *(const bf16x4*)p;
}

enum { EPI_HEADS = 0, EPI_OUT = 1 };

// ---------------------------------------------------------------------------
// Projection GEMM:  Y[m,n] = sum_k A[m,k] * W[n,k] + bias[n]
// A: [4096,1024] (f32 or bf16 row-major), W: [1024,1024] f32 row-major (NT).
// 128x128 tile / 256 threads (8 waves as 2x4, each wave 64x32 = 4x2 WMMA).
// K staged 64 wide (16 barrier rounds, 2 WMMA k-steps per round).
// EPI_HEADS: bf16 head layout [b,h,q,dk]. EPI_OUT: f32 row-major.
// ---------------------------------------------------------------------------
template<int EPI, typename AT>
__global__ __launch_bounds__(256) void gemm_proj_kernel(
    const AT* __restrict__ A, const float* __restrict__ W,
    const float* __restrict__ bias, void* __restrict__ dstv)
{
  __shared__ __align__(16) __bf16 As[128 * LDT64];
  __shared__ __align__(16) __bf16 Bs[128 * LDT64];
  const int tid  = threadIdx.x;
  const int lane = tid & 31;
  const int wid  = tid >> 5;
  const int m0   = blockIdx.y * 128;
  const int n0   = blockIdx.x * 128;
  const int wm0  = (wid >> 2) * 64;   // waves 2(M) x 4(N)
  const int wn0  = (wid & 3) * 32;

  v8f acc[4][2] = {};

  for (int kk = 0; kk < D_; kk += 64) {
    __syncthreads();
    // 128x64 panels, 4-elem chunks: 2048 chunks each, 8 per thread.
    #pragma unroll
    for (int i = 0; i < 8; ++i) {
      int c = tid + 256 * i;
      int r = c >> 4, koff = (c & 15) * 4;
      *(bf16x4*)&As[r * LDT64 + koff] = ld4_cvt(&A[(size_t)(m0 + r) * D_ + kk + koff]);
      *(bf16x4*)&Bs[r * LDT64 + koff] = ld4_cvt(&W[(size_t)(n0 + r) * D_ + kk + koff]);
    }
    if (kk + 64 < D_) {  // gfx1250 global_prefetch_b8 for the next K panel
      __builtin_prefetch(&A[(size_t)(m0 + (tid >> 1)) * D_ + kk + 64 + (tid & 1) * 32], 0, 0);
      __builtin_prefetch(&W[(size_t)(n0 + (tid >> 1)) * D_ + kk + 64 + (tid & 1) * 32], 0, 0);
    }
    __syncthreads();
    #pragma unroll
    for (int k0 = 0; k0 < 64; k0 += 32) {
      v16bf af[4], bfr[2];
      #pragma unroll
      for (int mt = 0; mt < 4; ++mt) af[mt] = load_frag(As, wm0 + mt * 16, k0, LDT64);
      #pragma unroll
      for (int nt = 0; nt < 2; ++nt) bfr[nt] = load_frag(Bs, wn0 + nt * 16, k0, LDT64);
      #pragma unroll
      for (int mt = 0; mt < 4; ++mt)
        #pragma unroll
        for (int nt = 0; nt < 2; ++nt)
          acc[mt][nt] = wmma_bf16(af[mt], bfr[nt], acc[mt][nt]);
    }
  }

  const int khalf = lane >> 4, nl = lane & 15;
  #pragma unroll
  for (int mt = 0; mt < 4; ++mt)
    #pragma unroll
    for (int nt = 0; nt < 2; ++nt)
      #pragma unroll
      for (int r = 0; r < 8; ++r) {
        int m = m0 + wm0 + mt * 16 + khalf * 8 + r;
        int n = n0 + wn0 + nt * 16 + nl;
        float v = acc[mt][nt][r] + bias[n];
        if (EPI == EPI_HEADS) {
          __bf16* dst = (__bf16*)dstv;
          int b = m >> 10, q = m & (N_ - 1);
          int h = n >> 6,  dk = n & (DK_ - 1);
          dst[(((size_t)(b * H_ + h)) * N_ + q) * DK_ + dk] = (__bf16)v;
        } else {
          ((float*)dstv)[(size_t)m * D_ + n] = v;
        }
      }
}

// ---------------------------------------------------------------------------
// rs[row, r] = Q[row,:] . rel_k_emb[r,:]   row in [0, BH*N), r in [0, 65)
// One wave per row; lanes cover r = lane, lane+32, lane+64.
// ---------------------------------------------------------------------------
__global__ __launch_bounds__(256) void rel_scores_kernel(
    const __bf16* __restrict__ Qh, const float* __restrict__ relk,
    float* __restrict__ rs)
{
  const int row  = blockIdx.x * 8 + (threadIdx.x >> 5);
  const int lane = threadIdx.x & 31;
  const __bf16* qp = Qh + (size_t)row * DK_;
  #pragma unroll
  for (int it = 0; it < 3; ++it) {
    int rr = lane + it * 32;
    if (rr < NR_) {
      const float* ep = relk + (size_t)rr * DK_;
      float acc = 0.f;
      #pragma unroll
      for (int dk = 0; dk < DK_; ++dk) acc += (float)qp[dk] * ep[dk];
      rs[(size_t)row * NR_ + rr] = acc;
    }
  }
}

// ---------------------------------------------------------------------------
// scores[bh,q,k] = (Q.K^T)/8 + rs[bh,q, clip(k-q)]   -> written to attn buf
// Per-(bh) batched NT GEMM, full K=64 panel staged once (bf16x8 loads),
// then 2 WMMA k-steps; 128x128 tiles.
// ---------------------------------------------------------------------------
__global__ __launch_bounds__(256) void scores_kernel(
    const __bf16* __restrict__ Qh, const __bf16* __restrict__ Kh,
    const float* __restrict__ rs, float* __restrict__ attn)
{
  __shared__ __align__(16) __bf16 As[128 * LDT64];
  __shared__ __align__(16) __bf16 Bs[128 * LDT64];
  const int bh = blockIdx.z;
  const __bf16* Qp = Qh + (size_t)bh * N_ * DK_;
  const __bf16* Kp = Kh + (size_t)bh * N_ * DK_;
  const int tid = threadIdx.x, lane = tid & 31, wid = tid >> 5;
  const int m0 = blockIdx.y * 128, n0 = blockIdx.x * 128;
  const int wm0 = (wid >> 2) * 64, wn0 = (wid & 3) * 32;

  // Stage whole 128x64 panels once: 1024 bf16x8 chunks each, 4 per thread.
  #pragma unroll
  for (int i = 0; i < 4; ++i) {
    int c = tid + 256 * i;
    int r = c >> 3, koff = (c & 7) * 8;
    *(bf16x8*)&As[r * LDT64 + koff] = *(const bf16x8*)&Qp[(size_t)(m0 + r) * DK_ + koff];
    *(bf16x8*)&Bs[r * LDT64 + koff] = *(const bf16x8*)&Kp[(size_t)(n0 + r) * DK_ + koff];
  }
  __syncthreads();

  v8f acc[4][2] = {};
  #pragma unroll
  for (int k0 = 0; k0 < DK_; k0 += 32) {
    v16bf af[4], bfr[2];
    #pragma unroll
    for (int mt = 0; mt < 4; ++mt) af[mt] = load_frag(As, wm0 + mt * 16, k0, LDT64);
    #pragma unroll
    for (int nt = 0; nt < 2; ++nt) bfr[nt] = load_frag(Bs, wn0 + nt * 16, k0, LDT64);
    #pragma unroll
    for (int mt = 0; mt < 4; ++mt)
      #pragma unroll
      for (int nt = 0; nt < 2; ++nt)
        acc[mt][nt] = wmma_bf16(af[mt], bfr[nt], acc[mt][nt]);
  }

  const int khalf = lane >> 4, nl = lane & 15;
  const float* rsb = rs + (size_t)bh * N_ * NR_;
  #pragma unroll
  for (int mt = 0; mt < 4; ++mt)
    #pragma unroll
    for (int nt = 0; nt < 2; ++nt)
      #pragma unroll
      for (int r = 0; r < 8; ++r) {
        int q = m0 + wm0 + mt * 16 + khalf * 8 + r;
        int kc = n0 + wn0 + nt * 16 + nl;
        int d = kc - q;
        d = d < -MRP_ ? -MRP_ : (d > MRP_ ? MRP_ : d);
        float v = acc[mt][nt][r] * 0.125f + rsb[(size_t)q * NR_ + d + MRP_];
        attn[((size_t)bh * N_ + q) * N_ + kc] = v;
      }
}

// ---------------------------------------------------------------------------
// In-LDS row softmax (row length 1024, float4 I/O) + rel-V bucket weights:
//   w[0] = sum_{k<=q-32} attn,  w[64] = sum_{k>=q+32} attn,
//   w[r] = attn[q, q+r-32] for r in 1..63 (0 if out of range).
// ---------------------------------------------------------------------------
__global__ __launch_bounds__(256) void softmax_kernel(
    float* __restrict__ attn, float* __restrict__ wbuf)
{
  __shared__ float rowv[N_];
  __shared__ float red[256];
  const int row = blockIdx.x;            // bh*N + q
  const int q   = row & (N_ - 1);
  const int tid = threadIdx.x;
  float* ap = attn + (size_t)row * N_;
  const int k4 = tid * 4;

  f32x4 v4 = *(const f32x4*)(ap + k4);
  float lmax = fmaxf(fmaxf(v4[0], v4[1]), fmaxf(v4[2], v4[3]));
  red[tid] = lmax; __syncthreads();
  for (int s = 128; s > 0; s >>= 1) {
    if (tid < s) red[tid] = fmaxf(red[tid], red[tid + s]);
    __syncthreads();
  }
  const float mx = red[0]; __syncthreads();

  float lsum = 0.f;
  #pragma unroll
  for (int j = 0; j < 4; ++j) {
    float e = __expf(v4[j] - mx);
    v4[j] = e; lsum += e;
  }
  red[tid] = lsum; __syncthreads();
  for (int s = 128; s > 0; s >>= 1) {
    if (tid < s) red[tid] += red[tid + s];
    __syncthreads();
  }
  const float inv = 1.f / red[0]; __syncthreads();

  float p0 = 0.f, p64 = 0.f;
  #pragma unroll
  for (int j = 0; j < 4; ++j) {
    float v = v4[j] * inv;
    v4[j] = v;
    rowv[k4 + j] = v;
    if (k4 + j <= q - MRP_) p0  += v;
    if (k4 + j >= q + MRP_) p64 += v;
  }
  *(f32x4*)(ap + k4) = v4;                       // final attn to d_out

  red[tid] = p0; __syncthreads();
  for (int s = 128; s > 0; s >>= 1) {
    if (tid < s) red[tid] += red[tid + s];
    __syncthreads();
  }
  const float w0 = red[0]; __syncthreads();
  red[tid] = p64; __syncthreads();
  for (int s = 128; s > 0; s >>= 1) {
    if (tid < s) red[tid] += red[tid + s];
    __syncthreads();
  }
  const float w64 = red[0]; __syncthreads();

  float* wp = wbuf + (size_t)row * NR_;
  if (tid == 0) { wp[0] = w0; wp[NR_ - 1] = w64; }
  if (tid >= 1 && tid <= 63) {
    int j = q + tid - MRP_;
    wp[tid] = (j >= 0 && j < N_) ? rowv[j] : 0.f;
  }
}

// ---------------------------------------------------------------------------
// ctx[bh,q,dk] = attn . V  +  w . rel_v_emb  (rel term fused as 3 extra
// zero-padded K-steps). NN GEMM: V/rel_v staged transposed into LDS [n][k].
// 128x64 tile, waves 4x2, each 32x32 = 2x2 WMMA. Output bf16 [b,q,h*64+dk].
// ---------------------------------------------------------------------------
__global__ __launch_bounds__(256) void ctx_kernel(
    const float* __restrict__ attn, const __bf16* __restrict__ Vh,
    const float* __restrict__ wbuf, const float* __restrict__ relv,
    __bf16* __restrict__ ctx)
{
  __shared__ __align__(16) __bf16 As[128 * LDT32];
  __shared__ __align__(16) __bf16 Bs[64 * LDT32];
  const int bh = blockIdx.z;
  const int b = bh >> 4, h = bh & (H_ - 1);
  const int tid = threadIdx.x, lane = tid & 31, wid = tid >> 5;
  const int m0  = blockIdx.x * 128;
  const int wm0 = (wid >> 1) * 32, wn0 = (wid & 1) * 32;
  const float*  ap = attn + (size_t)bh * N_ * N_;
  const __bf16* vp = Vh   + (size_t)bh * N_ * DK_;
  const float*  wp = wbuf + (size_t)bh * N_ * NR_;

  v8f acc[2][2] = {};
  for (int ks = 0; ks < 35; ++ks) {       // 32 steps attn.V + 3 steps w.relv
    __syncthreads();
    if (ks < 32) {
      const int kk = ks * 32;
      // attn panel 128x32: 1024 f32x4 chunks, 4 per thread.
      #pragma unroll
      for (int i = 0; i < 4; ++i) {
        int c = tid + 256 * i;
        int r = c >> 3, koff = (c & 7) * 4;
        *(bf16x4*)&As[r * LDT32 + koff] = ld4_cvt(&ap[(size_t)(m0 + r) * N_ + kk + koff]);
      }
      // V panel 32x64 -> LDS [n][k]: one bf16x8 load (coalesced over dk),
      // scalar DS scatter on the LDS side only.
      {
        int k = tid >> 3, noff = (tid & 7) * 8;
        bf16x8 v = *(const bf16x8*)&vp[(size_t)(kk + k) * DK_ + noff];
        #pragma unroll
        for (int j = 0; j < 8; ++j) Bs[(noff + j) * LDT32 + k] = v[j];
      }
      if (kk + 32 < N_)
        __builtin_prefetch(&ap[(size_t)(m0 + (tid >> 1)) * N_ + kk + 32 + (tid & 1) * 16], 0, 0);
    } else {
      const int kk = (ks - 32) * 32;
      #pragma unroll
      for (int e = tid; e < 128 * 32; e += 256) {
        int r = e >> 5, k = e & 31, rr = kk + k;
        As[r * LDT32 + k] = (rr < NR_) ? (__bf16)wp[(size_t)(m0 + r) * NR_ + rr]
                                       : (__bf16)0.f;
      }
      #pragma unroll
      for (int e = tid; e < 64 * 32; e += 256) {
        int n = e & 63, k = e >> 6, rr = kk + k;
        Bs[n * LDT32 + k] = (rr < NR_) ? (__bf16)relv[(size_t)rr * DK_ + n]
                                       : (__bf16)0.f;
      }
    }
    __syncthreads();
    v16bf af[2], bfr[2];
    #pragma unroll
    for (int mt = 0; mt < 2; ++mt) af[mt] = load_frag(As, wm0 + mt * 16, 0, LDT32);
    #pragma unroll
    for (int nt = 0; nt < 2; ++nt) bfr[nt] = load_frag(Bs, wn0 + nt * 16, 0, LDT32);
    #pragma unroll
    for (int mt = 0; mt < 2; ++mt)
      #pragma unroll
      for (int nt = 0; nt < 2; ++nt)
        acc[mt][nt] = wmma_bf16(af[mt], bfr[nt], acc[mt][nt]);
  }

  const int khalf = lane >> 4, nl = lane & 15;
  #pragma unroll
  for (int mt = 0; mt < 2; ++mt)
    #pragma unroll
    for (int nt = 0; nt < 2; ++nt)
      #pragma unroll
      for (int r = 0; r < 8; ++r) {
        int q  = m0 + wm0 + mt * 16 + khalf * 8 + r;
        int dk = wn0 + nt * 16 + nl;
        ctx[((size_t)b * N_ + q) * D_ + h * DK_ + dk] = (__bf16)acc[mt][nt][r];
      }
}

// ---------------------------------------------------------------------------
extern "C" void kernel_launch(void* const* d_in, const int* in_sizes, int n_in,
                              void* d_out, int out_size, void* d_ws, size_t ws_size,
                              hipStream_t stream)
{
  (void)in_sizes; (void)n_in; (void)out_size; (void)ws_size;
  const float* query = (const float*)d_in[0];
  const float* key   = (const float*)d_in[1];
  const float* value = (const float*)d_in[2];
  const float* Wq    = (const float*)d_in[3];
  const float* bq    = (const float*)d_in[4];
  const float* Wk    = (const float*)d_in[5];
  const float* bk    = (const float*)d_in[6];
  const float* Wv    = (const float*)d_in[7];
  const float* bv    = (const float*)d_in[8];
  const float* Wo    = (const float*)d_in[9];
  const float* bo    = (const float*)d_in[10];
  const float* relk  = (const float*)d_in[11];
  const float* relv  = (const float*)d_in[12];

  float* out  = (float*)d_out;                       // [B,N,D]
  float* attn = out + (size_t)B_ * N_ * D_;          // [B,H,N,N]

  char* ws = (char*)d_ws;
  size_t off = 0;
  auto alloc = [&](size_t bytes) {
    void* p = ws + off;
    off += bytes;
    off = (off + 255) & ~(size_t)255;
    return p;
  };
  __bf16* Qh   = (__bf16*)alloc((size_t)BH_ * N_ * DK_ * sizeof(__bf16));
  __bf16* Kh   = (__bf16*)alloc((size_t)BH_ * N_ * DK_ * sizeof(__bf16));
  __bf16* Vh   = (__bf16*)alloc((size_t)BH_ * N_ * DK_ * sizeof(__bf16));
  float*  rsB  = (float*) alloc((size_t)BH_ * N_ * NR_ * sizeof(float));
  float*  wB   = (float*) alloc((size_t)BH_ * N_ * NR_ * sizeof(float));
  __bf16* ctxB = (__bf16*)alloc((size_t)B_ * N_ * D_ * sizeof(__bf16));

  const dim3 blk(256);
  const dim3 gProj(D_ / 128, M_ / 128);              // (8, 32)

  gemm_proj_kernel<EPI_HEADS, float><<<gProj, blk, 0, stream>>>(query, Wq, bq, Qh);
  gemm_proj_kernel<EPI_HEADS, float><<<gProj, blk, 0, stream>>>(key,   Wk, bk, Kh);
  gemm_proj_kernel<EPI_HEADS, float><<<gProj, blk, 0, stream>>>(value, Wv, bv, Vh);

  rel_scores_kernel<<<BH_ * N_ / 8, blk, 0, stream>>>(Qh, relk, rsB);

  scores_kernel<<<dim3(N_ / 128, N_ / 128, BH_), blk, 0, stream>>>(Qh, Kh, rsB, attn);

  softmax_kernel<<<BH_ * N_, blk, 0, stream>>>(attn, wB);

  ctx_kernel<<<dim3(N_ / 128, 1, BH_), blk, 0, stream>>>(attn, Vh, wB, relv, ctxB);

  gemm_proj_kernel<EPI_OUT, __bf16><<<gProj, blk, 0, stream>>>(ctxB, Wo, bo, out);
}
